// Decoder_86809878987119
// MI455X (gfx1250) — matrix operand order repaired and checked by reading, not compiled, not run on hardware
//
#include <hip/hip_runtime.h>

#define B_SZ    32
#define ENC_T   1024
#define DEC_T   1024
#define DD      256
#define KW      31
#define PADW    15
#define KDIM    64
#define HALOSZ  (ENC_T + 2*PADW + 2)   // 1056
#define NTHREADS 256
#define MTILES  16                      // DD/16
#define NTILES  64                      // ENC_T/16
#define NWAVES  8
#define NT_PER_WAVE (NTILES/NWAVES)     // 8

typedef __attribute__((ext_vector_type(16))) __bf16 v16bf;
typedef __attribute__((ext_vector_type(8)))  __bf16 v8bf;
typedef __attribute__((ext_vector_type(8)))  float  v8f;
typedef __attribute__((ext_vector_type(4))) unsigned int u32x4;
typedef __attribute__((ext_vector_type(8))) int i32x8;
typedef __attribute__((ext_vector_type(4))) int i32x4;

#if __has_builtin(__builtin_amdgcn_tensor_load_to_lds) && __has_builtin(__builtin_amdgcn_s_wait_tensorcnt)
#define HAVE_TDM 1
#else
#define HAVE_TDM 0
#endif

// Native gfx1250 V_TANH_F32 if the builtin exists; otherwise branchless exp/rcp.
__device__ __forceinline__ float fast_tanh(float x) {
#if __has_builtin(__builtin_amdgcn_tanhf)
    return __builtin_amdgcn_tanhf(x);
#else
    const float xc = fminf(fmaxf(x, -10.f), 10.f);
    const float e2 = __expf(xc + xc);                 // v_exp_f32
    return (e2 - 1.f) * __builtin_amdgcn_rcpf(e2 + 1.f);
#endif
}

#if HAVE_TDM
// TDM: DMA one row of `n4` floats from global to LDS (descriptor per ISA 8.3/8.4).
__device__ __forceinline__ void tdm_load_row(const float* gsrc, unsigned lds_off, int n4) {
    const unsigned long long ga = (unsigned long long)(uintptr_t)gsrc;
    u32x4 g0;
    g0[0] = 1u;                                            // count=1, user mode
    g0[1] = lds_off;                                       // lds_addr (bytes)
    g0[2] = (unsigned)(ga & 0xFFFFFFFFull);                // global_addr[31:0]
    g0[3] = (unsigned)((ga >> 32) & 0x01FFFFFFull)         // global_addr[56:32]
          | (2u << 30);                                    // type=2 ("image")
    i32x8 g1;
    g1[0] = (int)(2u << 16);       // workgroup_mask=0, data_size=2 (4B), no extras
    g1[1] = (int)((unsigned)n4 << 16);  // tensor_dim0[15:0] in [31:16]
    g1[2] = (int)(1u << 16);       // tensor_dim0 hi=0 | tensor_dim1 lo16 = 1
    g1[3] = (int)((unsigned)n4 << 16);  // tensor_dim1 hi=0 | tile_dim0 = n4
    g1[4] = 1;                     // tile_dim1=1, tile_dim2=0
    g1[5] = n4;                    // tensor_dim0_stride lo32
    g1[6] = 0;
    g1[7] = 0;
    const i32x4 z4 = (i32x4){0, 0, 0, 0};
#if defined(__clang_major__) && __clang_major__ >= 23
    const i32x8 z8 = (i32x8){0, 0, 0, 0, 0, 0, 0, 0};
    __builtin_amdgcn_tensor_load_to_lds(g0, g1, z4, z4, z8, 0);
#else
    __builtin_amdgcn_tensor_load_to_lds(g0, g1, z4, z4, 0);
#endif
}
#endif

__global__ __launch_bounds__(NTHREADS)
void attn_decoder_kernel(const float* __restrict__ enc,      // [B, ENC_T, D]
                         const float* __restrict__ mel,      // [B, DEC_T, D]
                         const int*   __restrict__ enc_len,  // [B]
                         const int*   __restrict__ out_len,  // [B]
                         const float* __restrict__ wv,       // [1, D]
                         const float* __restrict__ convw,    // [D, 2, KW]
                         float* __restrict__ att_outs,       // [B, DEC_T, D]
                         float* __restrict__ att_scores)     // [B, DEC_T, ENC_T]
{
    __shared__ __bf16 s_A[DD * KDIM];      // conv weights, K padded to 64 (k=31,63 zero)
    __shared__ __bf16 s_pb[HALOSZ];        // prev score windows (bf16, halo)
    __shared__ __bf16 s_cb[HALOSZ];        // cum score windows (bf16, halo)
    __shared__ float  s_score[ENC_T];
    __shared__ float  s_cum[ENC_T];
    __shared__ float  s_energy[ENC_T];
    __shared__ float  s_bias[DD];          // mel_t*mask + att_out
    __shared__ float  s_wv[DD];
    __shared__ float  s_attout[DD];
    __shared__ float  s_red[NTHREADS];
    __shared__ float  s_melst[2][DD];      // TDM-staged mel rows (double buffered)

    const int b    = blockIdx.x;
    const int tid  = threadIdx.x;
    const int lane = tid & 31;
    const int wave = tid >> 5;
    const int tcol = lane & 15;
    const int hi   = (lane >= 16);
    const int klo  = hi ? 8 : 0;           // A-fragment K sub-offset (ISA A layout)
    const int klob = hi ? 8 : 0;           // B-fragment K sub-offset (table order, matches A)

    const float* encB = enc + (size_t)b * ENC_T * DD;
    const float* melB = mel + (size_t)b * DEC_T * DD;
    float* outB = att_outs   + (size_t)b * DEC_T * DD;
    float* scrB = att_scores + (size_t)b * DEC_T * ENC_T;
    const int e_len = enc_len[b];
    const int o_len = out_len[b];

    // ---- one-time init ----
    {
        const int d = tid;
        #pragma unroll 1
        for (int k = 0; k < KDIM; ++k) {
            float v = 0.f;
            if (k < KW)                      v = convw[d * 2 * KW + k];
            else if (k >= 32 && k < 32 + KW) v = convw[d * 2 * KW + KW + (k - 32)];
            s_A[d * KDIM + k] = (__bf16)v;
        }
        s_wv[d]     = wv[d];
        s_attout[d] = 0.f;
    }
    for (int i = tid; i < HALOSZ; i += NTHREADS) { s_pb[i] = (__bf16)0.f; s_cb[i] = (__bf16)0.f; }
    for (int i = tid; i < ENC_T;  i += NTHREADS) { s_cum[i] = 0.f; }
#if HAVE_TDM
    if (wave == 0) {
        tdm_load_row(melB, (unsigned)(uintptr_t)&s_melst[0][0], DD);   // stage step 0
        __builtin_amdgcn_s_wait_tensorcnt(0);
    }
#endif
    __syncthreads();

    // ---- sequential attention scan ----
    for (int step = 0; step < DEC_T; ++step) {
        // phase 1: bias = mel_t*mask + att_out ; clear energies ; kick next TDM
        const float dmask = (step < o_len) ? 1.f : 0.f;
#if HAVE_TDM
        const float mymel = s_melst[step & 1][tid];
        if (wave == 0 && step + 1 < DEC_T)
            tdm_load_row(melB + (size_t)(step + 1) * DD,
                         (unsigned)(uintptr_t)&s_melst[(step + 1) & 1][0], DD);
#else
        const float mymel = melB[(size_t)step * DD + tid];
        if (step + 1 < DEC_T)
            __builtin_prefetch(melB + (size_t)(step + 1) * DD + tid, 0, 1);
#endif
        s_bias[tid] = mymel * dmask + s_attout[tid];
        #pragma unroll
        for (int i = 0; i < ENC_T / NTHREADS; ++i) s_energy[tid + i * NTHREADS] = 0.f;
        __syncthreads();

        // phase 2: loc-conv as WMMA GEMM [256x64]x[64x1024] + native tanh + D-reduce
        // Hoist this wave's 8 N-tile B fragments (score windows) into registers.
        v16bf bP[NT_PER_WAVE], bC[NT_PER_WAVE];
        #pragma unroll
        for (int ni = 0; ni < NT_PER_WAVE; ++ni) {
            const int t = (wave * NT_PER_WAVE + ni) * 16 + tcol;
            const __bf16* pp = s_pb + t + klob;
            const __bf16* pc = s_cb + t + klob;
            #pragma unroll
            for (int j = 0; j < 16; ++j) { bP[ni][j] = pp[j < 8 ? j : j + 8]; bC[ni][j] = pc[j < 8 ? j : j + 8]; }
            if (hi) { bP[ni][15] = (__bf16)0.f; bC[ni][15] = (__bf16)0.f; }  // K=31/63 pad
        }

        float acc[NT_PER_WAVE];
        #pragma unroll
        for (int i = 0; i < NT_PER_WAVE; ++i) acc[i] = 0.f;

        #pragma unroll 1
        for (int mt = 0; mt < MTILES; ++mt) {
            const int M0   = mt * 16;
            const int mrow = M0 + tcol;
            // A fragments (kbase 0 / 32), 16B-aligned LDS vector loads
            const v8bf* pa = (const v8bf*)(s_A + mrow * KDIM);
            const v8bf l0 = pa[(0  + klo) >> 3];
            const v8bf h0 = pa[(16 + klo) >> 3];
            const v8bf l1 = pa[(32 + klo) >> 3];
            const v8bf h1 = pa[(48 + klo) >> 3];
            const v16bf afr0 = __builtin_shufflevector(l0, h0, 0,1,2,3,4,5,6,7,8,9,10,11,12,13,14,15);
            const v16bf afr1 = __builtin_shufflevector(l1, h1, 0,1,2,3,4,5,6,7,8,9,10,11,12,13,14,15);

            float biasr[8], wvr[8];
            const int dbase = M0 + (hi ? 8 : 0);
            #pragma unroll
            for (int r = 0; r < 8; ++r) { biasr[r] = s_bias[dbase + r]; wvr[r] = s_wv[dbase + r]; }

            #pragma unroll
            for (int ni = 0; ni < NT_PER_WAVE; ++ni) {
                v8f c = {};
                c = __builtin_amdgcn_wmma_f32_16x16x32_bf16(false, afr0, false, bP[ni],
                                                            (short)0, c, false, false);
                c = __builtin_amdgcn_wmma_f32_16x16x32_bf16(false, afr1, false, bC[ni],
                                                            (short)0, c, false, false);
                float e = 0.f;
                #pragma unroll
                for (int r = 0; r < 8; ++r)
                    e += wvr[r] * fast_tanh(biasr[r] + c[r]);
                acc[ni] += e;
            }
        }
        // two commutative ds_add_f32 per t (lanes L and L+16) -> bitwise deterministic
        #pragma unroll
        for (int ni = 0; ni < NT_PER_WAVE; ++ni) {
            const int t = (wave * NT_PER_WAVE + ni) * 16 + tcol;
            atomicAdd(&s_energy[t], acc[ni]);
        }
        __syncthreads();

        // phase 3: softmax over enc_T, pad-mask, update prev/cum windows, emit scores
        float le[4];
        float lmax = -3.0e38f;
        #pragma unroll
        for (int i = 0; i < 4; ++i) { le[i] = s_energy[tid + i * NTHREADS]; lmax = fmaxf(lmax, le[i]); }
        s_red[tid] = lmax; __syncthreads();
        for (int off = NTHREADS / 2; off > 0; off >>= 1) {
            if (tid < off) s_red[tid] = fmaxf(s_red[tid], s_red[tid + off]);
            __syncthreads();
        }
        const float gmax = s_red[0]; __syncthreads();
        float lsum = 0.f;
        #pragma unroll
        for (int i = 0; i < 4; ++i) { le[i] = __expf(le[i] - gmax); lsum += le[i]; }
        s_red[tid] = lsum; __syncthreads();
        for (int off = NTHREADS / 2; off > 0; off >>= 1) {
            if (tid < off) s_red[tid] += s_red[tid + off];
            __syncthreads();
        }
        const float inv = 1.f / s_red[0]; __syncthreads();

        float* scrRow = scrB + (size_t)step * ENC_T;
        #pragma unroll
        for (int i = 0; i < 4; ++i) {
            const int t  = tid + i * NTHREADS;
            const float sc = (t < e_len) ? le[i] * inv : 0.f;   // softmax then pad-zero (ref order)
            const float cu = s_cum[t] + sc;
            s_score[t] = sc;
            s_cum[t]   = cu;
            s_pb[PADW + t] = (__bf16)sc;
            s_cb[PADW + t] = (__bf16)cu;
            scrRow[t] = sc;
        }
        __syncthreads();

        // phase 4: att_out[d] = sum_t score[t] * enc[t,d]  (L2-resident GEMV)
        float o = 0.f;
        const float* ec = encB + tid;
        #pragma unroll 1
        for (int t = 0; t < ENC_T; t += 4) {
            o += s_score[t + 0] * ec[(size_t)(t + 0) * DD];
            o += s_score[t + 1] * ec[(size_t)(t + 1) * DD];
            o += s_score[t + 2] * ec[(size_t)(t + 2) * DD];
            o += s_score[t + 3] * ec[(size_t)(t + 3) * DD];
        }
        s_attout[tid] = o;                              // carry (unmasked, matches ref scan)
        outB[(size_t)step * DD + tid] = o * dmask;      // masked output

#if HAVE_TDM
        if (wave == 0 && step + 1 < DEC_T)
            __builtin_amdgcn_s_wait_tensorcnt(0);       // next mel row staged before reuse
#endif
        __syncthreads();
    }
}

extern "C" void kernel_launch(void* const* d_in, const int* in_sizes, int n_in,
                              void* d_out, int out_size, void* d_ws, size_t ws_size,
                              hipStream_t stream) {
    const float* enc   = (const float*)d_in[0];   // encoder_outputs [32,1024,256]
    const float* mel   = (const float*)d_in[1];   // melenc_outputs  [32,1024,256]
    const int*   elen  = (const int*)d_in[2];     // encoder_lengths [32]
    const int*   olen  = (const int*)d_in[3];     // output_lengths  [32]
    const float* wv    = (const float*)d_in[4];   // act_linear_w    [1,256]
    const float* convw = (const float*)d_in[5];   // location_conv_w [256,2,31]

    float* att_outs   = (float*)d_out;                               // [32,1024,256]
    float* att_scores = att_outs + (size_t)B_SZ * DEC_T * DD;        // [32,1024,1024]

    hipLaunchKernelGGL(attn_decoder_kernel, dim3(B_SZ), dim3(NTHREADS), 0, stream,
                       enc, mel, elen, olen, wv, convw, att_outs, att_scores);
}
